// SparseSelfAttention_13915694039402
// MI455X (gfx1250) — matrix operand order repaired and checked by reading, not compiled, not run on hardware
//
#include <hip/hip_runtime.h>
#include <stdint.h>

// ---------------------------------------------------------------------------
// MI455X / gfx1250 sparse self-attention
//   B=2, T=2048, D=1024, H=16, head_dim=64, TOPK=128, feature_keep=4
// Pipeline: f32->bf16 casts, 3x WMMA projection GEMMs (+top-4 feature
// sparsify), fused attention (scores->top-128->softmax->PV) with a full
// 16x2048 score stripe resident in LDS, final WMMA output GEMM (f32 out).
// Workspace use: 24M bf16 elements = 48 MB of d_ws.
// ---------------------------------------------------------------------------

typedef __bf16 bf16_t;
typedef __attribute__((ext_vector_type(16))) __bf16 v16bf;
typedef __attribute__((ext_vector_type(8)))  float  v8f;
typedef __attribute__((ext_vector_type(2)))  int    v2i_t;
typedef __attribute__((ext_vector_type(4)))  int    v4i_t;

// Only assert on the device pass; host pass has no AMDGPU builtins.
#if defined(__HIP_DEVICE_COMPILE__) && \
    !__has_builtin(__builtin_amdgcn_wmma_f32_16x16x32_bf16)
#error "gfx1250 WMMA bf16 builtin not available"
#endif

#define AS_GLOBAL __attribute__((address_space(1)))
#define AS_LDS    __attribute__((address_space(3)))

__device__ __forceinline__ void async_b64(void* lds_dst, const void* g_src) {
#if __has_builtin(__builtin_amdgcn_global_load_async_to_lds_b64)
  __builtin_amdgcn_global_load_async_to_lds_b64(
      (AS_GLOBAL v2i_t*)(void*)g_src, (AS_LDS v2i_t*)lds_dst, 0, 0);
#else
  *(uint64_t*)lds_dst = *(const uint64_t*)g_src;
#endif
}

__device__ __forceinline__ void async_b128(void* lds_dst, const void* g_src) {
#if __has_builtin(__builtin_amdgcn_global_load_async_to_lds_b128)
  __builtin_amdgcn_global_load_async_to_lds_b128(
      (AS_GLOBAL v4i_t*)(void*)g_src, (AS_LDS v4i_t*)lds_dst, 0, 0);
#else
  *(uint4*)lds_dst = *(const uint4*)g_src;
#endif
}

__device__ __forceinline__ void async_wait0() {
#if __has_builtin(__builtin_amdgcn_s_wait_asynccnt)
  __builtin_amdgcn_s_wait_asynccnt(0);
#else
  asm volatile("s_wait_asynccnt 0" ::: "memory");
#endif
}

constexpr int Bn = 2, Tn = 2048, Dn = 1024, Hn = 16, HDn = 64, TOPKn = 128;
constexpr size_t NX = (size_t)Bn * Tn * Dn;  // 4,194,304
constexpr size_t NW = (size_t)Dn * Dn;       // 1,048,576

// ---------------------------------------------------------------------------
__global__ void f32_to_bf16_kernel(const float* __restrict__ src,
                                   bf16_t* __restrict__ dst, int n) {
  int i = blockIdx.x * 256 + threadIdx.x;
  if (i < n) dst[i] = (bf16_t)src[i];
}

// ---------------------------------------------------------------------------
// Y[m, n0+j] = sum_k A[m,k]*W[n0+j,k], 16(M)x64(N) tile per WG, 4 waves.
// Epilogue: per row keep top-4 |v| of the 64 head features, bf16 out.
// ---------------------------------------------------------------------------
__global__ void proj_topk_kernel(const bf16_t* __restrict__ A,
                                 const bf16_t* __restrict__ W,
                                 bf16_t* __restrict__ Y) {
  __shared__ bf16_t sA[16][32];
  __shared__ bf16_t sB[64][32];
  __shared__ float  sO[16][64];

  const int tid  = threadIdx.x;      // 0..127
  const int wave = tid >> 5;         // 0..3 -> N-subtile
  const int lane = tid & 31;
  const int m0 = blockIdx.x * 16;
  const int n0 = blockIdx.y * 64;
  const int mrow = lane & 15, hi = lane >> 4;

  v8f acc = {};
  for (int kb = 0; kb < Dn; kb += 32) {
    {  // stage A tile: 16x32 bf16, 128 threads x 8B (async DMA to LDS)
      int r = tid >> 3, c = (tid & 7) * 4;
      async_b64(&sA[r][c], A + (size_t)(m0 + r) * Dn + kb + c);
    }
    {  // stage B tile: 64 rows of W x 32 k, 128 threads x 32B
      int r = tid >> 1, c = (tid & 1) * 16;
      const bf16_t* g = W + (size_t)(n0 + r) * Dn + kb + c;
      async_b128(&sB[r][c], g);
      async_b128(&sB[r][c + 8], g + 8);
      __builtin_prefetch(g + 32, 0, 0);  // next K-step -> global_prefetch_b8
    }
    async_wait0();
    __syncthreads();

    v16bf af, bfr;
#pragma unroll
    for (int v = 0; v < 8; ++v) {  // A frag: M=mrow, K pairs per ISA layout
      int k0 = ((v & 3) * 2) + ((v >> 2) * 16) + hi * 8;
      af[2 * v]     = sA[mrow][k0];
      af[2 * v + 1] = sA[mrow][k0 + 1];
    }
#pragma unroll
    for (int j = 0; j < 8; ++j) {  // B frag: N=wave*16+mrow, K pairs per ISA layout
      int k0 = ((j & 3) * 2) + ((j >> 2) * 16) + hi * 8;
      bfr[2 * j]     = sB[wave * 16 + mrow][k0];
      bfr[2 * j + 1] = sB[wave * 16 + mrow][k0 + 1];
    }
    acc = __builtin_amdgcn_wmma_f32_16x16x32_bf16(false, af, false, bfr,
                                                  (short)0, acc, false, false);
    __syncthreads();
  }

#pragma unroll
  for (int r = 0; r < 8; ++r)  // C/D layout: M=r+hi*8, N=lane&15
    sO[r + hi * 8][wave * 16 + mrow] = acc[r];
  __syncthreads();

  if (tid < 16) {  // per-token top-4 feature sparsify
    float b0 = -1.f, b1 = -1.f, b2 = -1.f, b3 = -1.f;
    for (int j = 0; j < 64; ++j) {
      float a = fabsf(sO[tid][j]);
      if (a > b3) {
        b3 = a;
        if (b3 > b2) { float t = b2; b2 = b3; b3 = t; }
        if (b2 > b1) { float t = b1; b1 = b2; b2 = t; }
        if (b1 > b0) { float t = b0; b0 = b1; b1 = t; }
      }
    }
    float thr = b3;
    for (int j = 0; j < 64; ++j) {
      float v = sO[tid][j];
      Y[(size_t)(m0 + tid) * Dn + n0 + j] =
          (bf16_t)((fabsf(v) >= thr) ? v : 0.f);
    }
  }
}

// ---------------------------------------------------------------------------
// Fused attention: one WG = (b, h, 16 query rows). 8 waves / 256 threads.
// Full 16x2048 f32 score stripe lives in LDS (128 KB of the 320 KB WGP LDS).
// ---------------------------------------------------------------------------
__global__ void attn_kernel(const bf16_t* __restrict__ Q,
                            const bf16_t* __restrict__ K,
                            const bf16_t* __restrict__ V,
                            bf16_t* __restrict__ O) {
  __shared__ float  sc[16][2048];      // 128 KB score stripe
  __shared__ bf16_t sq[16][64];        // Q tile
  __shared__ bf16_t sk[8][16][64];     // per-wave K staging
  __shared__ float  sOacc[8][16][16];  // per-wave PV partials
  __shared__ float  rowThr[16], rowMax[16], rowInv[16];

  const int tid  = threadIdx.x;  // 0..255
  const int wave = tid >> 5;     // 0..7
  const int lane = tid & 31;
  const int q0 = blockIdx.x * 16;
  const int h  = blockIdx.y;
  const int b  = blockIdx.z;
  const int mrow = lane & 15, hi = lane >> 4;
  const float scale = 0.125f;  // 1/sqrt(64)

  {  // load Q tile 16x64 bf16: 256 threads x 8B
    int r = tid >> 4, c = (tid & 15) * 4;
    async_b64(&sq[r][c], Q + ((size_t)b * Tn + q0 + r) * Dn + h * HDn + c);
  }
  async_wait0();
  __syncthreads();

  // ---- Phase 1: scores = Q K^T * scale, causal-masked, into LDS ----------
  for (int ct = wave; ct < Tn / 16; ct += 8) {
    const int kc = ct * 16;
    {  // stage 16 keys x 64 feats per wave: 32 lanes x 64B
      int r = lane >> 1, c = (lane & 1) * 32;
      const bf16_t* g = K + ((size_t)b * Tn + kc + r) * Dn + h * HDn + c;
      async_b128(&sk[wave][r][c], g);
      async_b128(&sk[wave][r][c + 8], g + 8);
      async_b128(&sk[wave][r][c + 16], g + 16);
      async_b128(&sk[wave][r][c + 24], g + 24);
    }
    async_wait0();  // wave-private buffer: per-wave wait suffices

    v8f acc = {};
#pragma unroll
    for (int half = 0; half < 2; ++half) {  // head_dim 64 = 2 x K32
      v16bf af, bfr;
#pragma unroll
      for (int v = 0; v < 8; ++v) {
        int k0 = ((v & 3) * 2) + ((v >> 2) * 16) + hi * 8 + half * 32;
        af[2 * v]     = sq[mrow][k0];
        af[2 * v + 1] = sq[mrow][k0 + 1];
      }
#pragma unroll
      for (int j = 0; j < 8; ++j) {  // B = K^T: N=key(mrow), Kdim=feat pairs per ISA layout
        int k0 = ((j & 3) * 2) + ((j >> 2) * 16) + hi * 8 + half * 32;
        bfr[2 * j]     = sk[wave][mrow][k0];
        bfr[2 * j + 1] = sk[wave][mrow][k0 + 1];
      }
      acc = __builtin_amdgcn_wmma_f32_16x16x32_bf16(false, af, false, bfr,
                                                    (short)0, acc, false, false);
    }
#pragma unroll
    for (int r = 0; r < 8; ++r) {
      int row = r + hi * 8;
      int col = kc + mrow;
      float s = acc[r] * scale;
      sc[row][col] = (col <= q0 + row) ? s : -__builtin_inff();
    }
  }
  __syncthreads();

  // ---- Phase 2: per-row top-128 threshold + softmax constants ------------
  for (int rr = wave; rr < 16; rr += 8) {
    const int q = q0 + rr;
    float mx = -__builtin_inff();
    for (int c = lane; c < Tn; c += 32) mx = fmaxf(mx, sc[rr][c]);
#pragma unroll
    for (int o = 16; o > 0; o >>= 1) mx = fmaxf(mx, __shfl_xor(mx, o, 32));

    float thr;
    if (q + 1 <= TOPKn) {
      thr = -__builtin_inff();  // fewer valid entries than k: keep all
    } else {
      float lo = -1.0e5f, hiB = mx;  // binary search the 128th largest
      for (int it = 0; it < 24; ++it) {
        float mid = 0.5f * (lo + hiB);
        int cnt = 0;
        for (int c = lane; c < Tn; c += 32) cnt += (sc[rr][c] > mid) ? 1 : 0;
#pragma unroll
        for (int o = 16; o > 0; o >>= 1) cnt += __shfl_xor(cnt, o, 32);
        if (cnt >= TOPKn) lo = mid; else hiB = mid;
      }
      thr = lo;
    }
    float sm = 0.f;
    for (int c = lane; c < Tn; c += 32) {
      float s = sc[rr][c];
      if (s >= thr) sm += __expf(s - mx);
    }
#pragma unroll
    for (int o = 16; o > 0; o >>= 1) sm += __shfl_xor(sm, o, 32);
    if (lane == 0) {
      rowThr[rr] = thr;
      rowMax[rr] = mx;
      rowInv[rr] = 1.0f / sm;
    }
  }
  __syncthreads();

  // ---- Phase 3: O = P V ; P built on the fly from the LDS stripe ---------
  const int nt = wave & 3, n0 = nt * 16;  // feature subtile
  const int kh = wave >> 2;               // K-dim interleave (2-way)
  const int kend = q0 + 16;               // all-zero P beyond the diagonal
  const float thrv = rowThr[mrow], mxv = rowMax[mrow], inv = rowInv[mrow];

  v8f acc = {};
  const int nsteps = (kend + 31) / 32;
  for (int s = kh; s < nsteps; s += 2) {
    const int kb = s * 32;
    v16bf af, bfr;
#pragma unroll
    for (int v = 0; v < 8; ++v) {  // P fragment (A): exp/normalize inline
      int k0 = kb + ((v & 3) * 2) + ((v >> 2) * 16) + hi * 8;
      float s0 = sc[mrow][k0];
      float s1 = sc[mrow][k0 + 1];
      af[2 * v]     = (bf16_t)((s0 >= thrv) ? __expf(s0 - mxv) * inv : 0.f);
      af[2 * v + 1] = (bf16_t)((s1 >= thrv) ? __expf(s1 - mxv) * inv : 0.f);
    }
#pragma unroll
    for (int j = 0; j < 8; ++j) {  // V fragment (B): N=feat(mrow), K=key pairs per ISA layout
      int k0 = kb + ((j & 3) * 2) + ((j >> 2) * 16) + hi * 8;
      size_t vb = ((size_t)b * Tn + k0) * Dn + h * HDn + n0 + mrow;
      bfr[2 * j]     = V[vb];
      bfr[2 * j + 1] = V[vb + Dn];
    }
    acc = __builtin_amdgcn_wmma_f32_16x16x32_bf16(false, af, false, bfr,
                                                  (short)0, acc, false, false);
  }
#pragma unroll
  for (int r = 0; r < 8; ++r) sOacc[wave][r + hi * 8][mrow] = acc[r];
  __syncthreads();

  for (int e = tid; e < 16 * 64; e += 256) {  // combine K-halves, emit bf16
    int r = e >> 6, c = e & 63;
    int tile = c >> 4, cc = c & 15;
    float val = sOacc[tile][r][cc] + sOacc[tile + 4][r][cc];
    O[((size_t)b * Tn + q0 + r) * Dn + h * HDn + c] = (bf16_t)val;
  }
}

// ---------------------------------------------------------------------------
// out = attn_out @ Wo^T, dense f32 output.
// ---------------------------------------------------------------------------
__global__ void gemm_out_kernel(const bf16_t* __restrict__ A,
                                const bf16_t* __restrict__ W,
                                float* __restrict__ Y) {
  __shared__ bf16_t sA[16][32];
  __shared__ bf16_t sB[64][32];
  __shared__ float  sO[16][64];

  const int tid  = threadIdx.x;
  const int wave = tid >> 5;
  const int lane = tid & 31;
  const int m0 = blockIdx.x * 16;
  const int n0 = blockIdx.y * 64;
  const int mrow = lane & 15, hi = lane >> 4;

  v8f acc = {};
  for (int kb = 0; kb < Dn; kb += 32) {
    {
      int r = tid >> 3, c = (tid & 7) * 4;
      async_b64(&sA[r][c], A + (size_t)(m0 + r) * Dn + kb + c);
    }
    {
      int r = tid >> 1, c = (tid & 1) * 16;
      const bf16_t* g = W + (size_t)(n0 + r) * Dn + kb + c;
      async_b128(&sB[r][c], g);
      async_b128(&sB[r][c + 8], g + 8);
    }
    async_wait0();
    __syncthreads();

    v16bf af, bfr;
#pragma unroll
    for (int v = 0; v < 8; ++v) {
      int k0 = ((v & 3) * 2) + ((v >> 2) * 16) + hi * 8;
      af[2 * v]     = sA[mrow][k0];
      af[2 * v + 1] = sA[mrow][k0 + 1];
    }
#pragma unroll
    for (int j = 0; j < 8; ++j) {
      int k0 = ((j & 3) * 2) + ((j >> 2) * 16) + hi * 8;
      bfr[2 * j]     = sB[wave * 16 + mrow][k0];
      bfr[2 * j + 1] = sB[wave * 16 + mrow][k0 + 1];
    }
    acc = __builtin_amdgcn_wmma_f32_16x16x32_bf16(false, af, false, bfr,
                                                  (short)0, acc, false, false);
    __syncthreads();
  }

#pragma unroll
  for (int r = 0; r < 8; ++r) sO[r + hi * 8][wave * 16 + mrow] = acc[r];
  __syncthreads();

  {
    int r = tid >> 3, c0 = (tid & 7) * 8;
    for (int j = 0; j < 8; ++j)
      Y[(size_t)(m0 + r) * Dn + n0 + c0 + j] = sO[r][c0 + j];
  }
}

// ---------------------------------------------------------------------------
extern "C" void kernel_launch(void* const* d_in, const int* in_sizes, int n_in,
                              void* d_out, int out_size, void* d_ws,
                              size_t ws_size, hipStream_t stream) {
  const float* x  = (const float*)d_in[0];
  const float* Wq = (const float*)d_in[1];
  const float* Wk = (const float*)d_in[2];
  const float* Wv = (const float*)d_in[3];
  const float* Wo = (const float*)d_in[4];
  float* out = (float*)d_out;

  bf16_t* p = (bf16_t*)d_ws;  // 48 MB of workspace used
  bf16_t* xbf = p; p += NX;
  bf16_t* wqb = p; p += NW;
  bf16_t* wkb = p; p += NW;
  bf16_t* wvb = p; p += NW;
  bf16_t* wob = p; p += NW;
  bf16_t* qbf = p; p += NX;
  bf16_t* kbf = p; p += NX;
  bf16_t* vbf = p; p += NX;
  bf16_t* obf = p; p += NX;

  f32_to_bf16_kernel<<<dim3((unsigned)((NX + 255) / 256)), 256, 0, stream>>>(x, xbf, (int)NX);
  f32_to_bf16_kernel<<<dim3((unsigned)((NW + 255) / 256)), 256, 0, stream>>>(Wq, wqb, (int)NW);
  f32_to_bf16_kernel<<<dim3((unsigned)((NW + 255) / 256)), 256, 0, stream>>>(Wk, wkb, (int)NW);
  f32_to_bf16_kernel<<<dim3((unsigned)((NW + 255) / 256)), 256, 0, stream>>>(Wv, wvb, (int)NW);
  f32_to_bf16_kernel<<<dim3((unsigned)((NW + 255) / 256)), 256, 0, stream>>>(Wo, wob, (int)NW);

  dim3 gp((Bn * Tn) / 16, Dn / 64);  // 256 x 16
  proj_topk_kernel<<<gp, 128, 0, stream>>>(xbf, wqb, qbf);
  proj_topk_kernel<<<gp, 128, 0, stream>>>(xbf, wkb, kbf);
  proj_topk_kernel<<<gp, 128, 0, stream>>>(xbf, wvb, vbf);

  attn_kernel<<<dim3(Tn / 16, Hn, Bn), 256, 0, stream>>>(qbf, kbf, vbf, obf);

  gemm_out_kernel<<<gp, 128, 0, stream>>>(obf, wob, out);
}
